// MultiHeadAttention_876173328459
// MI455X (gfx1250) — compile-verified, matrix operand order checked by reading, not run on hardware
//
#include <hip/hip_runtime.h>

typedef __attribute__((ext_vector_type(16))) _Float16 v16h;
typedef __attribute__((ext_vector_type(8)))  _Float16 v8h;
typedef __attribute__((ext_vector_type(8)))  float    v8f;
typedef int v4i __attribute__((vector_size(16)));

#define WMMA_F16(a, b, c) \
  __builtin_amdgcn_wmma_f32_16x16x32_f16(false, (a), false, (b), (short)0, (c), false, false)

#if defined(__HIP_DEVICE_COMPILE__) && __has_builtin(__builtin_amdgcn_global_load_async_to_lds_b128)
#define HAVE_ASYNC_LDS 1
#else
#define HAVE_ASYNC_LDS 0
#endif

// 16-byte global -> LDS copy: async DMA on CDNA5, plain round-trip otherwise.
static __device__ __forceinline__ void cp16(_Float16* ldst, const _Float16* gsrc) {
#if HAVE_ASYNC_LDS
  __builtin_amdgcn_global_load_async_to_lds_b128(
      (__attribute__((address_space(1))) v4i*)(gsrc),
      (__attribute__((address_space(3))) v4i*)(ldst), 0, 0);
#else
  *(v8h*)ldst = *(const v8h*)gsrc;
#endif
}

static __device__ __forceinline__ void async_join() {
#if HAVE_ASYNC_LDS
#if __has_builtin(__builtin_amdgcn_s_wait_asynccnt)
  __builtin_amdgcn_s_wait_asynccnt(0);
#else
  asm volatile("s_wait_asynccnt 0" ::: "memory");
#endif
#endif
}

static __device__ __forceinline__ v16h join8(v8h lo, v8h hi) {
  union { v16h v; v8h h[2]; } u;
  u.h[0] = lo; u.h[1] = hi;
  return u.v;
}

// ---------------------------------------------------------------------------
// fp32 -> fp16 conversion (memory-bound, negligible)
// ---------------------------------------------------------------------------
__global__ void cvt_f32_f16(const float* __restrict__ in,
                            _Float16* __restrict__ out, int n) {
  int i = blockIdx.x * blockDim.x + threadIdx.x;
  int stride = gridDim.x * blockDim.x;
  for (; i < n; i += stride) out[i] = (_Float16)in[i];
}

// fp32 [rows,cols] -> fp16 transposed [cols,rows]; 32x32 LDS tile
__global__ __launch_bounds__(256)
void cvt_transpose(const float* __restrict__ in, _Float16* __restrict__ out,
                   int rows, int cols) {
  __shared__ float tile[32][33];
  int c0 = blockIdx.x * 32, r0 = blockIdx.y * 32;
  int tx = threadIdx.x, ty = threadIdx.y;   // (32, 8)
#pragma unroll
  for (int j = 0; j < 32; j += 8)
    tile[ty + j][tx] = in[(size_t)(r0 + ty + j) * cols + c0 + tx];
  __syncthreads();
#pragma unroll
  for (int j = 0; j < 32; j += 8)
    out[(size_t)(c0 + ty + j) * rows + r0 + tx] = (_Float16)tile[tx][ty + j];
}

// ---------------------------------------------------------------------------
// WMMA GEMM: C[M,N] = A[M,K] * Bt[N,K]^T + bias   (A, Bt fp16 row-major)
// block = 256 threads (8 waves), block tile 128x256, wave tile 64x64, KB = 32,
// double-buffered LDS with async global->LDS staging.
// MODE 0: fp16 out; MODE 1: fp32 out; MODE 2: fused QKV out (Q,K normal fp16,
//         V written transposed into vt[b*16+h][d][n]).
// ---------------------------------------------------------------------------
template <int MODE>
__global__ __launch_bounds__(256)
void gemm_f16_wmma(const _Float16* __restrict__ A, const _Float16* __restrict__ Bt,
                   const float* __restrict__ bias,
                   _Float16* __restrict__ Oh, float* __restrict__ Of,
                   _Float16* __restrict__ vt,
                   int M, int N, int K) {
  __shared__ _Float16 Al[2][128 * 40];   // [row][k], stride 40 halfs (80B)
  __shared__ _Float16 Bl[2][256 * 40];   // [col][k]

  const int tid  = threadIdx.x;
  const int lane = tid & 31;
  const int wave = tid >> 5;
  const int wm = wave >> 2;              // 0..1
  const int wn = wave & 3;               // 0..3
  const int l15 = lane & 15;
  const int hf  = lane >> 4;
  const int mBlk = blockIdx.y * 128;
  const int nBlk = blockIdx.x * 256;
  const int steps = K >> 5;

  auto stage = [&](int it) {
    int k0 = it << 5;
    _Float16* Ab = Al[it & 1];
    _Float16* Bb = Bl[it & 1];
#pragma unroll
    for (int t = 0; t < 2; ++t) {        // A tile: 128 rows x 32 k
      int idx = tid + t * 256;
      int r = idx >> 2, ch = idx & 3;
      cp16(Ab + r * 40 + ch * 8, A + (size_t)(mBlk + r) * K + k0 + ch * 8);
    }
#pragma unroll
    for (int t = 0; t < 4; ++t) {        // B tile: 256 cols x 32 k (from Bt rows)
      int idx = tid + t * 256;
      int r = idx >> 2, ch = idx & 3;
      cp16(Bb + r * 40 + ch * 8, Bt + (size_t)(nBlk + r) * K + k0 + ch * 8);
    }
  };

  v8f acc[4][4] = {};

  stage(0);
  for (int it = 0; it < steps; ++it) {
    async_join();
    __syncthreads();
    if (it + 1 < steps) stage(it + 1);   // DMA overlaps the WMMAs below

    const _Float16* Ab = Al[it & 1];
    const _Float16* Bb = Bl[it & 1];
    v16h af[4];
#pragma unroll
    for (int fr = 0; fr < 4; ++fr) {
      const _Float16* p = Ab + (wm * 64 + fr * 16 + l15) * 40 + hf * 8;
      af[fr] = join8(*(const v8h*)p, *(const v8h*)(p + 16));
    }
#pragma unroll
    for (int fc = 0; fc < 4; ++fc) {
      const _Float16* p = Bb + (wn * 64 + fc * 16 + l15) * 40 + hf * 8;
      v16h bf = join8(*(const v8h*)p, *(const v8h*)(p + 16));
#pragma unroll
      for (int fr = 0; fr < 4; ++fr)
        acc[fr][fc] = WMMA_F16(af[fr], bf, acc[fr][fc]);
    }
  }

  // epilogue: C/D layout -> row = frag*16 + 8*hf + i, col = base + l15
#pragma unroll
  for (int fc = 0; fc < 4; ++fc) {
    int col = nBlk + wn * 64 + fc * 16 + l15;
    float bv = bias ? bias[col] : 0.f;
#pragma unroll
    for (int fr = 0; fr < 4; ++fr) {
      int rbase = mBlk + wm * 64 + fr * 16 + hf * 8;
#pragma unroll
      for (int i = 0; i < 8; ++i) {
        float v = acc[fr][fc][i] + bv;
        int row = rbase + i;
        if (MODE == 1) {
          Of[(size_t)row * N + col] = v;
        } else if (MODE == 2 && col >= 2048) {
          // V -> transposed layout vt[(b*16+h)*64 + d][n], n = token in seq
          int hd = col - 2048;
          vt[((size_t)(row >> 11) * 1024 + hd) * 2048 + (row & 2047)] = (_Float16)v;
        } else {
          Oh[(size_t)row * N + col] = (_Float16)v;
        }
      }
    }
  }
}

// ---------------------------------------------------------------------------
// Flash attention. qkv: fp16 [B*N, 3072] (Q cols 0..1023, K cols 1024..2047),
// vt: fp16 [(b*16+h)*64 + d][n] (V pre-transposed by the QKV GEMM).
// grid = (N/256, B*H); block = 128 threads (4 waves); wave = 64 query rows.
// Double-buffered async K/V staging. Softmax scale folded into Q (2^-3 is
// exact in fp16), so S comes out of the WMMA pre-scaled.
// ---------------------------------------------------------------------------
__global__ __launch_bounds__(128)
void flash_attn_wmma(const _Float16* __restrict__ qkv,
                     const _Float16* __restrict__ vt,
                     _Float16* __restrict__ out) {
  constexpr int SEQ = 2048, C3 = 3072, COUT = 1024;
  __shared__ _Float16 Kl[2][64 * 72];     // [key][d]
  __shared__ _Float16 Vl[2][64 * 72];     // [d][key]
  __shared__ _Float16 Pl[4][64 * 72];     // per-wave P tile [qrow][key]

  const int tid  = threadIdx.x;
  const int lane = tid & 31;
  const int wave = tid >> 5;
  const int l15  = lane & 15;
  const int hf   = lane >> 4;
  const int b = blockIdx.y >> 4;
  const int h = blockIdx.y & 15;
  const int qBase = blockIdx.x * 256 + wave * 64;
  const size_t rowQ = (size_t)b * SEQ + qBase;

  auto stageKV = [&](int it) {
    int kb = it << 6;
    _Float16* Kb = Kl[it & 1];
    _Float16* Vb = Vl[it & 1];
#pragma unroll
    for (int t = 0; t < 4; ++t) {
      int idx = tid + t * 128;
      int r = idx >> 3, ch = idx & 7;     // r: key row / d row, ch: 8-half chunk
      cp16(Kb + r * 72 + ch * 8,
           qkv + ((size_t)b * SEQ + kb + r) * C3 + 1024 + h * 64 + ch * 8);
      cp16(Vb + r * 72 + ch * 8,
           vt + ((size_t)blockIdx.y * 64 + r) * SEQ + kb + ch * 8);
    }
  };

  // Q fragments (A layout), pre-scaled by HEAD_DIM^-0.5 = 0.125 (exact in f16)
  v16h qf[4][2];
#pragma unroll
  for (int fr = 0; fr < 4; ++fr)
#pragma unroll
    for (int ks = 0; ks < 2; ++ks) {
      const _Float16* p = qkv + (rowQ + fr * 16 + l15) * C3 + h * 64 + ks * 32 + hf * 8;
      v8h lo = *(const v8h*)p * (_Float16)0.125f;
      v8h hi = *(const v8h*)(p + 16) * (_Float16)0.125f;
      qf[fr][ks] = join8(lo, hi);
    }

  v8f O[4][4] = {};
  float mrun[4][8], lrun[4][8];
#pragma unroll
  for (int fr = 0; fr < 4; ++fr)
#pragma unroll
    for (int i = 0; i < 8; ++i) { mrun[fr][i] = -1e30f; lrun[fr][i] = 0.f; }

  stageKV(0);
  for (int it = 0; it < SEQ / 64; ++it) {
    async_join();
    __syncthreads();
    if (it + 1 < SEQ / 64) stageKV(it + 1);   // DMA overlaps compute

    const _Float16* Kb = Kl[it & 1];
    const _Float16* Vb = Vl[it & 1];

    // S = (Q*scale) * K^T  (B frag: lane = key col n, elements = d chunks)
    v8f S[4][4];
#pragma unroll
    for (int fc = 0; fc < 4; ++fc) {
      const _Float16* p0 = Kb + (fc * 16 + l15) * 72 + hf * 8;
      v16h kf0 = join8(*(const v8h*)p0,        *(const v8h*)(p0 + 16));
      v16h kf1 = join8(*(const v8h*)(p0 + 32), *(const v8h*)(p0 + 48));
#pragma unroll
      for (int fr = 0; fr < 4; ++fr) {
        v8f s = {};
        s = WMMA_F16(qf[fr][0], kf0, s);
        S[fr][fc] = WMMA_F16(qf[fr][1], kf1, s);
      }
    }

    // online softmax: row r = fr*16 + 8*hf + i; 16 cols/frag across lanes
#pragma unroll
    for (int fr = 0; fr < 4; ++fr) {
#pragma unroll
      for (int i = 0; i < 8; ++i) {
        float mx = -1e30f;
#pragma unroll
        for (int fc = 0; fc < 4; ++fc) mx = fmaxf(mx, S[fr][fc][i]);
#pragma unroll
        for (int m = 1; m <= 8; m <<= 1) mx = fmaxf(mx, __shfl_xor(mx, m));
        float mo = mrun[fr][i];
        float mn = fmaxf(mo, mx);
        float alpha = __expf(mo - mn);
        float rs = 0.f;
#pragma unroll
        for (int fc = 0; fc < 4; ++fc) {
          float p = __expf(S[fr][fc][i] - mn);
          rs += p;
          Pl[wave][(fr * 16 + hf * 8 + i) * 72 + fc * 16 + l15] = (_Float16)p;
        }
#pragma unroll
        for (int m = 1; m <= 8; m <<= 1) rs += __shfl_xor(rs, m);
        mrun[fr][i] = mn;
        lrun[fr][i] = lrun[fr][i] * alpha + rs;
#pragma unroll
        for (int fc = 0; fc < 4; ++fc) O[fr][fc][i] *= alpha;
      }
    }

    // reload P as A-fragments (wave-local LDS round trip does the layout swap)
    v16h pf[4][2];
#pragma unroll
    for (int fr = 0; fr < 4; ++fr)
#pragma unroll
      for (int ks = 0; ks < 2; ++ks) {
        const _Float16* p = Pl[wave] + (fr * 16 + l15) * 72 + ks * 32 + hf * 8;
        pf[fr][ks] = join8(*(const v8h*)p, *(const v8h*)(p + 16));
      }

    // O += P * V  (B frag from transposed Vl: lane = d col, elements = keys)
#pragma unroll
    for (int fc = 0; fc < 4; ++fc) {
      const _Float16* p0 = Vb + (fc * 16 + l15) * 72 + hf * 8;
      v16h vf0 = join8(*(const v8h*)p0,        *(const v8h*)(p0 + 16));
      v16h vf1 = join8(*(const v8h*)(p0 + 32), *(const v8h*)(p0 + 48));
#pragma unroll
      for (int fr = 0; fr < 4; ++fr) {
        O[fr][fc] = WMMA_F16(pf[fr][0], vf0, O[fr][fc]);
        O[fr][fc] = WMMA_F16(pf[fr][1], vf1, O[fr][fc]);
      }
    }
  }

  // epilogue: normalize and store to [B,N,H,D] layout = [8192,1024]
#pragma unroll
  for (int fc = 0; fc < 4; ++fc) {
    int d = fc * 16 + l15;
#pragma unroll
    for (int fr = 0; fr < 4; ++fr)
#pragma unroll
      for (int i = 0; i < 8; ++i) {
        int tok = qBase + fr * 16 + hf * 8 + i;
        float v = O[fr][fc][i] / lrun[fr][i];
        out[((size_t)b * SEQ + tok) * COUT + h * 64 + d] = (_Float16)v;
      }
  }
}

// ---------------------------------------------------------------------------
extern "C" void kernel_launch(void* const* d_in, const int* in_sizes, int n_in,
                              void* d_out, int out_size, void* d_ws, size_t ws_size,
                              hipStream_t stream) {
  const float* x      = (const float*)d_in[0];
  const float* qkv_w  = (const float*)d_in[1];
  const float* qkv_b  = (const float*)d_in[2];
  const float* proj_w = (const float*)d_in[3];
  const float* proj_b = (const float*)d_in[4];
  float* out = (float*)d_out;

  const int C = 1024, M = 4 * 2048;   // 8192 tokens

  char* ws = (char*)d_ws;
  _Float16* xh    = (_Float16*)ws; ws += (size_t)M * C * 2;          // 16 MiB
  _Float16* wqkvT = (_Float16*)ws; ws += (size_t)C * 3 * C * 2;      //  6 MiB  [3072,1024]
  _Float16* qkvh  = (_Float16*)ws; ws += (size_t)M * 3 * C * 2;      // 48 MiB
  _Float16* vt    = (_Float16*)ws; ws += (size_t)M * C * 2;          // 16 MiB  [64*64, 2048]
  _Float16* attnh = (_Float16*)ws; ws += (size_t)M * C * 2;          // 16 MiB
  _Float16* wpjT  = (_Float16*)ws;                                   //  2 MiB  [1024,1024]

  cvt_f32_f16<<<2048, 256, 0, stream>>>(x, xh, M * C);
  cvt_transpose<<<dim3(96, 32), dim3(32, 8), 0, stream>>>(qkv_w,  wqkvT, C, 3 * C);
  cvt_transpose<<<dim3(32, 32), dim3(32, 8), 0, stream>>>(proj_w, wpjT,  C, C);

  // QKV: [8192,1024] x [1024,3072] -> fp16 Q,K rows + transposed V
  gemm_f16_wmma<2><<<dim3(12, 64), 256, 0, stream>>>(
      xh, wqkvT, qkv_b, qkvh, nullptr, vt, M, 3 * C, C);

  // attention: grid (qblocks=8, B*H=64)
  flash_attn_wmma<<<dim3(8, 64), 128, 0, stream>>>(qkvh, vt, attnh);

  // projection: [8192,1024] x [1024,1024] -> fp32 out
  gemm_f16_wmma<1><<<dim3(4, 64), 256, 0, stream>>>(
      attnh, wpjT, proj_b, nullptr, out, nullptr, M, C, C);
}
